// Attention2ViewPillarNet_7765300871562
// MI455X (gfx1250) — matrix operand order repaired and checked by reading, not compiled, not run on hardware
//
#include <hip/hip_runtime.h>

#define BN_EPS 1e-3f

typedef __attribute__((ext_vector_type(16))) _Float16 v16h;
typedef __attribute__((ext_vector_type(8)))  _Float16 v8h;
typedef __attribute__((ext_vector_type(8)))  float    v8f;

#define kB  2
#define kN  30000
#define kP  (kB * kN)
#define XNX 432
#define XNY 496
#define XV  (XNX * XNY)      /* 214272 voxels per batch (nz==1) */
#define CNX 2560
#define CNY 100
#define CV  (CNX * CNY)      /* 256000 */

// ---------------------------------------------------------------------------
// BN folding: scale = g/sqrt(v+eps), bias = b - m*scale
__global__ void k_prep_sb(const float* __restrict__ g, const float* __restrict__ b,
                          const float* __restrict__ m, const float* __restrict__ v,
                          float* __restrict__ scale, float* __restrict__ bias, int C) {
  int i = blockIdx.x * blockDim.x + threadIdx.x;
  if (i < C) {
    float s = g[i] * rsqrtf(v[i] + BN_EPS);
    scale[i] = s;
    bias[i]  = b[i] - m[i] * s;
  }
}

// ---------------------------------------------------------------------------
__device__ __forceinline__ void vox_one(
    int m, int b, const float q[3], float msk,
    float vs0, float vs1, float vs2, float o0, float o1, float o2,
    int d0, int d1, int d2, int Vg,
    float* __restrict__ frac, float* __restrict__ center,
    int* __restrict__ idxA, float* __restrict__ padA,
    float* __restrict__ cnt, float* __restrict__ sum) {
  float vs[3] = {vs0, vs1, vs2};
  float of[3] = {o0, o1, o2};
  int   dd[3] = {d0, d1, d2};
  int c[3];
  bool in = true;
#pragma unroll
  for (int k = 0; k < 3; ++k) {
    float fr = (q[k] - of[k]) / vs[k];
    float cf = floorf(fr);
    c[k] = (int)cf;
    in = in && (c[k] >= 0) && (c[k] < dd[k]);
    frac[m * 3 + k]   = fr;
    center[m * 3 + k] = (cf + 0.5f) * vs[k] + of[k];
  }
  float pad = (msk >= 1.0f && in) ? 0.0f : 1.0f;
  int idx = (pad > 0.0f) ? 0 : (c[0] * (dd[1] * dd[2]) + c[1] * dd[2] + c[2]);
  idxA[m] = idx;
  padA[m] = pad;
  if (pad == 0.0f) {
    size_t vb = (size_t)b * Vg + idx;
    atomicAdd(&cnt[vb], 1.0f);
    atomicAdd(&sum[vb * 3 + 0], q[0]);
    atomicAdd(&sum[vb * 3 + 1], q[1]);
    atomicAdd(&sum[vb * 3 + 2], q[2]);
  }
}

__global__ void k_voxelize(const float* __restrict__ pxyz, const float* __restrict__ pmask,
                           float* xy_frac, float* xy_center, int* xy_idx, float* xy_pad,
                           float* xy_cnt, float* xy_sum,
                           float* cyl_p, float* cyl_frac, float* cyl_center, int* cyl_idx,
                           float* cyl_pad, float* cyl_cnt, float* cyl_sum) {
  int m = blockIdx.x * blockDim.x + threadIdx.x;
  if (m >= kP) return;
  int b = m / kN;
  float p[3] = {pxyz[m * 3 + 0], pxyz[m * 3 + 1], pxyz[m * 3 + 2]};
  float msk = pmask[m];
  // XY view
  vox_one(m, b, p, msk, 69.12f / 432.0f, 79.36f / 496.0f, 4.0f,
          0.0f, -39.68f, -3.0f, XNX, XNY, 1, XV,
          xy_frac, xy_center, xy_idx, xy_pad, xy_cnt, xy_sum);
  // Cylinder view
  float rho = sqrtf(p[0] * p[0] + p[1] * p[1]);
  float phi = atan2f(p[1], p[0]);
  float q[3] = {phi, p[2], rho};
  cyl_p[m * 3 + 0] = q[0];
  cyl_p[m * 3 + 1] = q[1];
  cyl_p[m * 3 + 2] = q[2];
  const float PI = 3.14159265358979323846f;
  vox_one(m, b, q, msk, (2.0f * PI) / 2560.0f, 4.0f / 100.0f, 69.12f,
          -PI, -3.0f, 0.0f, CNX, CNY, 1, CV,
          cyl_frac, cyl_center, cyl_idx, cyl_pad, cyl_cnt, cyl_sum);
}

__global__ void k_div_mean(float* __restrict__ sum, const float* __restrict__ cnt, int nV) {
  int i = blockIdx.x * blockDim.x + threadIdx.x;
  if (i < nV * 3) sum[i] /= fmaxf(cnt[i / 3], 1.0f);
}

__global__ void k_cov(const float* __restrict__ q, const int* __restrict__ idxA,
                      const float* __restrict__ padA, const float* __restrict__ mean,
                      float* __restrict__ cov, float* __restrict__ centered, int Vg) {
  int m = blockIdx.x * blockDim.x + threadIdx.x;
  if (m >= kP) return;
  int b = m / kN;
  if (padA[m] > 0.0f) {
    centered[m * 3 + 0] = 0.0f;
    centered[m * 3 + 1] = 0.0f;
    centered[m * 3 + 2] = 0.0f;
    return;
  }
  size_t vb = (size_t)b * Vg + idxA[m];
  float d[3];
#pragma unroll
  for (int k = 0; k < 3; ++k) {
    d[k] = q[m * 3 + k] - mean[vb * 3 + k];
    centered[m * 3 + k] = d[k];
  }
#pragma unroll
  for (int i = 0; i < 3; ++i)
#pragma unroll
    for (int j = 0; j < 3; ++j) atomicAdd(&cov[vb * 9 + i * 3 + j], d[i] * d[j]);
}

__global__ void k_div_cov(float* __restrict__ cov, const float* __restrict__ cnt, int nV) {
  int i = blockIdx.x * blockDim.x + threadIdx.x;
  if (i < nV * 9) cov[i] /= fmaxf(cnt[i / 9], 1.0f);
}

// 45 features, zero-padded to 64 columns so the GEMM K-dim is a clean multiple of 32
__global__ void k_feat(const float* __restrict__ pxyz, const float* __restrict__ pfeat,
                       const float* __restrict__ xy_center, const int* __restrict__ xy_idx,
                       const float* __restrict__ xy_cnt, const float* __restrict__ cen_xy,
                       const float* __restrict__ xy_cov, const float* __restrict__ xy_mean,
                       const float* __restrict__ cyl_p, const float* __restrict__ cyl_center,
                       const int* __restrict__ cyl_idx, const float* __restrict__ cyl_cnt,
                       const float* __restrict__ cen_cyl, const float* __restrict__ cyl_cov,
                       const float* __restrict__ cyl_mean, float* __restrict__ feat) {
  int m = blockIdx.x * blockDim.x + threadIdx.x;
  if (m >= kP) return;
  int b = m / kN;
  float* f = feat + (size_t)m * 64;
  size_t vx = (size_t)b * XV + xy_idx[m];
  size_t vc = (size_t)b * CV + cyl_idx[m];
#pragma unroll
  for (int d = 0; d < 3; ++d) {
    f[0 + d]  = pxyz[m * 3 + d];
    f[3 + d]  = pxyz[m * 3 + d] - xy_center[m * 3 + d];
    f[7 + d]  = cen_xy[m * 3 + d];
    f[19 + d] = xy_mean[vx * 3 + d];
    f[22 + d] = cyl_p[m * 3 + d];
    f[25 + d] = cyl_p[m * 3 + d] - cyl_center[m * 3 + d];
    f[29 + d] = cen_cyl[m * 3 + d];
    f[41 + d] = cyl_mean[vc * 3 + d];
  }
  f[6]  = xy_cnt[vx];
  f[28] = cyl_cnt[vc];
#pragma unroll
  for (int k = 0; k < 9; ++k) {
    f[10 + k] = xy_cov[vx * 9 + k];
    f[32 + k] = cyl_cov[vc * 9 + k];
  }
  f[44] = pfeat[m];
#pragma unroll
  for (int k = 45; k < 64; ++k) f[k] = 0.0f;
}

// ---------------------------------------------------------------------------
// Fused GEMM + BN + ReLU (+ optional row mask):  Y[m][n] = relu(X[m][:]·W[n][:]*s[n]+b[n])
// 256 threads = 8 waves/block; each wave computes a 16(M)x64(N) tile via four
// v_wmma_f32_16x16x32_f16 per K-step.  W (f32, [64][K]) staged to LDS as f16
// padded to Kpad.  A-fragments loaded straight from global X (f32,[M][Kpad]).
__global__ void k_gemm_bn_relu(const float* __restrict__ X, const float* __restrict__ W,
                               const float* __restrict__ scale, const float* __restrict__ bias,
                               const float* __restrict__ rowmask, float* __restrict__ Y,
                               int M, int K, int Kpad) {
  extern __shared__ _Float16 sW[];  // [64][Kpad]
  for (int t = threadIdx.x; t < 64 * Kpad; t += blockDim.x) {
    int o = t / Kpad, k = t - o * Kpad;
    sW[t] = (k < K) ? (_Float16)W[o * K + k] : (_Float16)0.0f;
  }
  __syncthreads();

  int wave = threadIdx.x >> 5;
  int lane = threadIdx.x & 31;
  int m0 = (blockIdx.x * 8 + wave) * 16;
  if (m0 >= M) return;
  int half = lane >> 4;    // lane-half selects K sub-blocks
  int lrow = lane & 15;    // A row / B,C,D column

  v8f acc[4] = {};
  for (int kb = 0; kb < Kpad; kb += 32) {
    // A fragment: row m0+lrow, elements K = kb + {0,16} + half*8 + 0..7
    const float* ap = X + (size_t)(m0 + lrow) * Kpad + kb + half * 8;
    v8f alo = *(const v8f*)ap;
    v8f ahi = *(const v8f*)(ap + 16);
    v16h a;
#pragma unroll
    for (int e = 0; e < 8; ++e) {
      a[e]     = (_Float16)alo[e];
      a[e + 8] = (_Float16)ahi[e];
    }
#pragma unroll
    for (int j = 0; j < 4; ++j) {
      // B fragment: col j*16+lrow, elements K = kb + {0,16} + half*8 + 0..7
      const _Float16* bp = sW + (size_t)(j * 16 + lrow) * Kpad + kb + half * 8;
      v8h b0 = *(const v8h*)bp;
      v8h b1 = *(const v8h*)(bp + 16);
      v16h bf;
#pragma unroll
      for (int e = 0; e < 8; ++e) {
        bf[e]     = b0[e];
        bf[e + 8] = b1[e];
      }
      acc[j] = __builtin_amdgcn_wmma_f32_16x16x32_f16(false, a, false, bf,
                                                      (short)0, acc[j], false, false);
    }
  }
#pragma unroll
  for (int j = 0; j < 4; ++j) {
    int n = j * 16 + lrow;
    float sc = scale[n], bi = bias[n];
#pragma unroll
    for (int r = 0; r < 8; ++r) {
      int mm = m0 + half * 8 + r;
      if (mm < M) {
        float v = fmaxf(acc[j][r] * sc + bi, 0.0f);
        if (rowmask) v *= rowmask[mm];
        Y[(size_t)mm * 64 + n] = v;
      }
    }
  }
}

// ---------------------------------------------------------------------------
// row-max (over 64 ch) -> y[m];  column-max (over points, per batch) -> z[b][c]
__global__ void k_rowcol_max(const float* __restrict__ X, float* __restrict__ y,
                             unsigned* __restrict__ z) {
  __shared__ unsigned sz[2 * 64];
  for (int t = threadIdx.x; t < 128; t += blockDim.x) sz[t] = 0u;
  __syncthreads();
  int m = blockIdx.x * blockDim.x + threadIdx.x;
  if (m < kP) {
    int b = m / kN;
    const float* xr = X + (size_t)m * 64;
    float mx = xr[0];
#pragma unroll 4
    for (int c = 0; c < 64; ++c) {
      float v = xr[c];
      mx = fmaxf(mx, v);
      atomicMax(&sz[b * 64 + c], __float_as_uint(v));  // post-ReLU => non-negative
    }
    y[m] = mx;
  }
  __syncthreads();
  for (int t = threadIdx.x; t < 128; t += blockDim.x) atomicMax(&z[t], sz[t]);
}

// t[b][i] = relu( sum_n y[b][n]*pa_w1[i][n] + pa_b1[i] ),  i<4
__global__ void k_pa(const float* __restrict__ y, const float* __restrict__ w1,
                     const float* __restrict__ b1, float* __restrict__ t4) {
  __shared__ float red[4 * 256];
  int b = blockIdx.x;
  float s[4] = {0.f, 0.f, 0.f, 0.f};
  for (int n = threadIdx.x; n < kN; n += blockDim.x) {
    float yv = y[b * kN + n];
#pragma unroll
    for (int i = 0; i < 4; ++i) s[i] += yv * w1[i * kN + n];
  }
#pragma unroll
  for (int i = 0; i < 4; ++i) red[i * 256 + threadIdx.x] = s[i];
  __syncthreads();
  if (threadIdx.x < 4) {
    float acc = 0.f;
    for (int t = 0; t < 256; ++t) acc += red[threadIdx.x * 256 + t];
    t4[b * 4 + threadIdx.x] = fmaxf(acc + b1[threadIdx.x], 0.0f);
  }
}

// ca[b][c] = relu(z@ca_w1.T + b1) @ ca_w2.T + b2
__global__ void k_ca(const float* __restrict__ z, const float* __restrict__ w1,
                     const float* __restrict__ b1, const float* __restrict__ w2,
                     const float* __restrict__ b2, float* __restrict__ cab) {
  __shared__ float h[4];
  int b = blockIdx.x;
  if (threadIdx.x < 4) {
    float a = 0.f;
    for (int c = 0; c < 64; ++c) a += z[b * 64 + c] * w1[threadIdx.x * 64 + c];
    h[threadIdx.x] = fmaxf(a + b1[threadIdx.x], 0.0f);
  }
  __syncthreads();
  int c = threadIdx.x;
  if (c < 64) {
    float a = 0.f;
#pragma unroll
    for (int j = 0; j < 4; ++j) a += h[j] * w2[c * 4 + j];
    cab[b * 64 + c] = a + b2[c];
  }
}

// out = x1 * sigmoid(pa(n) * ca(c))
__global__ void k_apply(const float* __restrict__ x1, const float* __restrict__ t4,
                        const float* __restrict__ paw2, const float* __restrict__ pab2,
                        const float* __restrict__ cab, float* __restrict__ out) {
  int t = blockIdx.x * blockDim.x + threadIdx.x;
  if (t >= kP * 64) return;
  int m = t >> 6, c = t & 63;
  int b = m / kN, n = m - b * kN;
  float pa = t4[b * 4 + 0] * paw2[n * 4 + 0] + t4[b * 4 + 1] * paw2[n * 4 + 1] +
             t4[b * 4 + 2] * paw2[n * 4 + 2] + t4[b * 4 + 3] * paw2[n * 4 + 3] + pab2[n];
  float a = pa * cab[b * 64 + c];
  float s = 1.0f / (1.0f + __expf(-a));
  out[t] = x1[t] * s;
}

// segment-max scatter into voxel grid (values >=0 => u32 monotone compare)
__global__ void k_scatter_max(const float* __restrict__ xf, const int* __restrict__ idxA,
                              const float* __restrict__ padA, unsigned* __restrict__ vox,
                              int Vg) {
  int t = blockIdx.x * blockDim.x + threadIdx.x;
  if (t >= kP * 64) return;
  int m = t >> 6, c = t & 63;
  if (padA[m] != 0.0f) return;
  int b = m / kN;
  size_t o = ((size_t)b * Vg + idxA[m]) * 64 + c;
  atomicMax(&vox[o], __float_as_uint(xf[t]));
}

// bilinear gather from conv image F[b][ix*ny+iy][c] at fractional voxel coords
__global__ void k_bilinear(const float* __restrict__ F, const float* __restrict__ frac,
                           float* __restrict__ outp, int nx, int ny, int Vg) {
  int t = blockIdx.x * blockDim.x + threadIdx.x;
  if (t >= kP * 64) return;
  int m = t >> 6, c = t & 63, b = m / kN;
  float fx = frac[m * 3 + 0], fy = frac[m * 3 + 1];
  float x0f = floorf(fx), y0f = floorf(fy);
  int x0 = (int)x0f; x0 = x0 < 0 ? 0 : (x0 > nx - 1 ? nx - 1 : x0);
  int x1 = x0 + 1;   x1 = x1 > nx - 1 ? nx - 1 : x1;
  int y0 = (int)y0f; y0 = y0 < 0 ? 0 : (y0 > ny - 1 ? ny - 1 : y0);
  int y1 = y0 + 1;   y1 = y1 > ny - 1 ? ny - 1 : y1;
  float wx1 = fx - x0f, wx0 = 1.0f - wx1;
  float wy1 = fy - y0f, wy0 = 1.0f - wy1;
  const float* Fb = F + (size_t)b * Vg * 64;
  float v = Fb[((size_t)x0 * ny + y0) * 64 + c] * (wx0 * wy0) +
            Fb[((size_t)x0 * ny + y1) * 64 + c] * (wx0 * wy1) +
            Fb[((size_t)x1 * ny + y0) * 64 + c] * (wx1 * wy0) +
            Fb[((size_t)x1 * ny + y1) * 64 + c] * (wx1 * wy1);
  outp[t] = v;
}

__global__ void k_concat(const float* __restrict__ a, const float* __restrict__ b2,
                         const float* __restrict__ c3, float* __restrict__ cat) {
  int t = blockIdx.x * blockDim.x + threadIdx.x;
  if (t >= kP * 192) return;
  int m = t / 192, k = t - m * 192;
  float v = (k < 64) ? a[(size_t)m * 64 + k]
                     : (k < 128) ? b2[(size_t)m * 64 + (k - 64)]
                                 : c3[(size_t)m * 64 + (k - 128)];
  cat[t] = v;
}

// final pillar segment-max directly into transposed output [B,64,496,432]
__global__ void k_scatter_out(const float* __restrict__ x3, const int* __restrict__ xy_idx,
                              const float* __restrict__ xy_pad, unsigned* __restrict__ out) {
  int t = blockIdx.x * blockDim.x + threadIdx.x;
  if (t >= kP * 64) return;
  int m = t >> 6, c = t & 63;
  if (xy_pad[m] != 0.0f) return;
  int b = m / kN;
  int idx = xy_idx[m];
  int ix = idx / XNY, iy = idx - ix * XNY;
  size_t o = (((size_t)b * 64 + c) * XNY + iy) * XNX + ix;
  atomicMax(&out[o], __float_as_uint(x3[t]));
}

// ---------------------------------------------------------------------------
extern "C" void kernel_launch(void* const* d_in, const int* in_sizes, int n_in,
                              void* d_out, int out_size, void* d_ws, size_t ws_size,
                              hipStream_t stream) {
  (void)in_sizes; (void)n_in; (void)ws_size;
  int ii = 0;
  auto nextf = [&]() { return (const float*)d_in[ii++]; };
  const float* pxyz  = nextf();
  const float* pfeat = nextf();
  const float* pmask = nextf();
  struct PN { const float *w,*g,*b,*m,*v,*paw1,*pab1,*paw2,*pab2,*caw1,*cab1,*caw2,*cab2; };
  auto getPN = [&]() {
    PN p;
    p.w = nextf(); p.g = nextf(); p.b = nextf(); p.m = nextf(); p.v = nextf();
    p.paw1 = nextf(); p.pab1 = nextf(); p.paw2 = nextf(); p.pab2 = nextf();
    p.caw1 = nextf(); p.cab1 = nextf(); p.caw2 = nextf(); p.cab2 = nextf();
    return p;
  };
  PN pn1 = getPN(), pn2 = getPN(), pn3 = getPN(), xypn = getPN(), cylpn = getPN();
  struct RS { const float *w,*g,*b,*m,*v; };
  auto getRS = [&]() { RS r; r.w = nextf(); r.g = nextf(); r.b = nextf(); r.m = nextf(); r.v = nextf(); return r; };
  RS xyr = getRS(), cylr = getRS();

  const size_t VXB = (size_t)kB * XV;      // 428544
  const size_t VCB = (size_t)kB * CV;      // 512000
  const size_t P64 = (size_t)kP * 64 * 4;

  char* ws = (char*)d_ws;
  size_t off = 0;
  auto alloc = [&](size_t bytes) -> char* {
    char* p = ws + off;
    off = (off + bytes + 255) & ~(size_t)255;
    return p;
  };
  auto sub = [](size_t& o, size_t bytes) -> size_t {
    size_t r = o;
    o = (o + bytes + 255) & ~(size_t)255;
    return r;
  };

  // BIG region: phase 1 = voxel stats + feat; phase 2 = voxel-max + conv + x1
  size_t s1 = 0;
  size_t o_xy_cnt  = sub(s1, VXB * 4);
  size_t o_xy_mean = sub(s1, VXB * 12);
  size_t o_xy_cov  = sub(s1, VXB * 36);
  size_t o_cy_cnt  = sub(s1, VCB * 4);
  size_t o_cy_mean = sub(s1, VCB * 12);
  size_t o_cy_cov  = sub(s1, VCB * 36);
  size_t stats_end = s1;
  size_t o_featb   = sub(s1, P64);
  size_t s2 = 0;
  size_t o_vox = sub(s2, VCB * 64 * 4);
  size_t o_F   = sub(s2, VCB * 64 * 4);
  size_t o_x1  = sub(s2, P64);
  char* big = alloc(s1 > s2 ? s1 : s2);

  float* xy_cnt  = (float*)(big + o_xy_cnt);
  float* xy_mean = (float*)(big + o_xy_mean);
  float* xy_cov  = (float*)(big + o_xy_cov);
  float* cy_cnt  = (float*)(big + o_cy_cnt);
  float* cy_mean = (float*)(big + o_cy_mean);
  float* cy_cov  = (float*)(big + o_cy_cov);
  float* featb   = (float*)(big + o_featb);
  float* voxbuf  = (float*)(big + o_vox);
  float* Fbuf    = (float*)(big + o_F);
  float* x1      = (float*)(big + o_x1);

  // persistent per-point / small buffers
  float* xy_frac    = (float*)alloc((size_t)kP * 12);
  float* xy_center  = (float*)alloc((size_t)kP * 12);
  int*   xy_idx     = (int*)  alloc((size_t)kP * 4);
  float* xy_pad     = (float*)alloc((size_t)kP * 4);
  float* cen_xy     = (float*)alloc((size_t)kP * 12);
  float* cyl_p      = (float*)alloc((size_t)kP * 12);
  float* cyl_frac   = (float*)alloc((size_t)kP * 12);
  float* cyl_center = (float*)alloc((size_t)kP * 12);
  int*   cyl_idx    = (int*)  alloc((size_t)kP * 4);
  float* cyl_pad    = (float*)alloc((size_t)kP * 4);
  float* cen_cyl    = (float*)alloc((size_t)kP * 12);
  float* xbuf   = (float*)alloc(P64);
  float* xfbuf  = (float*)alloc(P64);
  float* xxy    = (float*)alloc(P64);
  float* xcyl   = (float*)alloc(P64);
  float* xpw    = (float*)alloc(P64);
  float* catbuf = (float*)alloc((size_t)kP * 192 * 4);
  float* ybuf   = (float*)alloc((size_t)kP * 4);
  float* zbuf   = (float*)alloc(128 * 4);
  float* t4buf  = (float*)alloc(8 * 4);
  float* cabuf  = (float*)alloc(128 * 4);
  float* sb     = (float*)alloc(7 * 128 * 4);  // folded BN per layer

  // fold BN for: 0=pn1 1=pn2 2=pn3 3=xy_pn 4=cyl_pn 5=xy_res1 6=cyl_res1
  const float* Lg[7] = {pn1.g, pn2.g, pn3.g, xypn.g, cylpn.g, xyr.g, cylr.g};
  const float* Lb[7] = {pn1.b, pn2.b, pn3.b, xypn.b, cylpn.b, xyr.b, cylr.b};
  const float* Lm[7] = {pn1.m, pn2.m, pn3.m, xypn.m, cylpn.m, xyr.m, cylr.m};
  const float* Lv[7] = {pn1.v, pn2.v, pn3.v, xypn.v, cylpn.v, xyr.v, cylr.v};
  for (int i = 0; i < 7; ++i)
    k_prep_sb<<<1, 64, 0, stream>>>(Lg[i], Lb[i], Lm[i], Lv[i], sb + i * 128, sb + i * 128 + 64, 64);

  const int gP   = (kP + 255) / 256;
  const int g64  = (int)(((size_t)kP * 64 + 255) / 256);

  // --- voxelization + stats ---
  hipMemsetAsync(big, 0, stats_end, stream);
  k_voxelize<<<gP, 256, 0, stream>>>(pxyz, pmask,
      xy_frac, xy_center, xy_idx, xy_pad, xy_cnt, xy_mean,
      cyl_p, cyl_frac, cyl_center, cyl_idx, cyl_pad, cy_cnt, cy_mean);
  k_div_mean<<<(int)((VXB * 3 + 255) / 256), 256, 0, stream>>>(xy_mean, xy_cnt, (int)VXB);
  k_div_mean<<<(int)((VCB * 3 + 255) / 256), 256, 0, stream>>>(cy_mean, cy_cnt, (int)VCB);
  k_cov<<<gP, 256, 0, stream>>>(pxyz,  xy_idx,  xy_pad,  xy_mean, xy_cov, cen_xy,  XV);
  k_cov<<<gP, 256, 0, stream>>>(cyl_p, cyl_idx, cyl_pad, cy_mean, cy_cov, cen_cyl, CV);
  k_div_cov<<<(int)((VXB * 9 + 255) / 256), 256, 0, stream>>>(xy_cov, xy_cnt, (int)VXB);
  k_div_cov<<<(int)((VCB * 9 + 255) / 256), 256, 0, stream>>>(cy_cov, cy_cnt, (int)VCB);
  k_feat<<<gP, 256, 0, stream>>>(pxyz, pfeat,
      xy_center, xy_idx, xy_cnt, cen_xy, xy_cov, xy_mean,
      cyl_p, cyl_center, cyl_idx, cy_cnt, cen_cyl, cy_cov, cy_mean, featb);

  auto run_pointnet = [&](const PN& p, int layer, const float* X, int K, int Kpad, float* out) {
    const float* sc = sb + layer * 128;
    k_gemm_bn_relu<<<(kP + 127) / 128, 256, 64 * Kpad * 2, stream>>>(
        X, p.w, sc, sc + 64, pmask, x1, kP, K, Kpad);
    hipMemsetAsync(zbuf, 0, 128 * 4, stream);
    k_rowcol_max<<<gP, 256, 0, stream>>>(x1, ybuf, (unsigned*)zbuf);
    k_pa<<<kB, 256, 0, stream>>>(ybuf, p.paw1, p.pab1, t4buf);
    k_ca<<<kB, 64, 0, stream>>>(zbuf, p.caw1, p.cab1, p.caw2, p.cab2, cabuf);
    k_apply<<<g64, 256, 0, stream>>>(x1, t4buf, p.paw2, p.pab2, cabuf, out);
  };

  auto run_view = [&](const PN& vp, int vlayer, const RS& rs, int rlayer,
                      int nx, int ny, int Vg, const int* idxA, const float* padA,
                      const float* fracA, float* outv) {
    run_pointnet(vp, vlayer, xbuf, 64, 64, xfbuf);
    hipMemsetAsync(voxbuf, 0, (size_t)kB * Vg * 64 * 4, stream);
    k_scatter_max<<<g64, 256, 0, stream>>>(xfbuf, idxA, padA, (unsigned*)voxbuf, Vg);
    const float* sc = sb + rlayer * 128;
    k_gemm_bn_relu<<<(int)(((size_t)kB * Vg + 127) / 128), 256, 64 * 64 * 2, stream>>>(
        voxbuf, rs.w, sc, sc + 64, nullptr, Fbuf, kB * Vg, 64, 64);
    k_bilinear<<<g64, 256, 0, stream>>>(Fbuf, fracA, outv, nx, ny, Vg);
  };

  // --- network ---
  run_pointnet(pn1, 0, featb, 45, 64, xbuf);                                   // x = pointnet(feat)
  run_view(xypn, 3, xyr, 5, XNX, XNY, XV, xy_idx, xy_pad, xy_frac, xxy);       // x_xy
  run_view(cylpn, 4, cylr, 6, CNX, CNY, CV, cyl_idx, cyl_pad, cyl_frac, xcyl); // x_cyl
  run_pointnet(pn2, 1, xbuf, 64, 64, xpw);                                     // x_pw
  k_concat<<<(int)(((size_t)kP * 192 + 255) / 256), 256, 0, stream>>>(xxy, xcyl, xpw, catbuf);
  run_pointnet(pn3, 2, catbuf, 192, 192, x1);                                  // in-place apply OK

  hipMemsetAsync(d_out, 0, (size_t)out_size * 4, stream);
  k_scatter_out<<<g64, 256, 0, stream>>>(x1, xy_idx, xy_pad, (unsigned*)d_out);
}
